// SelfAttention_56891136803403
// MI455X (gfx1250) — compile-verified, matrix operand order checked by reading, not run on hardware
//
#include <hip/hip_runtime.h>

typedef __attribute__((ext_vector_type(16))) _Float16 v16h;
typedef __attribute__((ext_vector_type(8)))  _Float16 v8h;
typedef __attribute__((ext_vector_type(4)))  _Float16 v4h;
typedef __attribute__((ext_vector_type(8)))  float    v8f;

#define IN_D   1024
#define OUT_D  128
#define NQ     8192
#define QSCALE 0.08838834764831845f  // 1/sqrt(128)

union FragA { v16h v; v8h h[2]; };

// Row-major f16 fragment loader (works for global or LDS generic pointers).
// A-fragment: lane = (m, g); halves 0-7 -> K = kb+8g..+7, halves 8-15 -> K = kb+16+8g..+7.
// B-fragment: identical mapping with lane index = N (matrix accessed "N-major").
__device__ __forceinline__ v16h load_frag_rm(const _Float16* __restrict__ base,
                                             int stride, int kb, int lane) {
  int m = lane & 15, g = lane >> 4;
  const _Float16* p = base + m * stride + kb + 8 * g;
  FragA u;
  u.h[0] = *(const v8h*)(p);
  u.h[1] = *(const v8h*)(p + 16);
  return u.v;
}

__device__ __forceinline__ v8f zero8() {
  v8f r;
#pragma unroll
  for (int i = 0; i < 8; ++i) r[i] = 0.0f;
  return r;
}

// CDNA5 async copy: global -> LDS, tracked by ASYNCcnt (inline asm: portable
// across ROCm7.2/clang-22 and amdgpu-toolchain/clang-23 builtin arity skew).
__device__ __forceinline__ void async_load_b128(unsigned lds_addr, const void* gptr) {
  asm volatile("global_load_async_to_lds_b128 %0, %1, off"
               :: "v"(lds_addr), "v"((unsigned long long)gptr)
               : "memory");
}
__device__ __forceinline__ void wait_asynccnt_le8() {
  asm volatile("s_wait_asynccnt 0x8" ::: "memory");
}

// ---------------------------------------------------------------- converts
__global__ void cvt_z_kernel(const float* __restrict__ z, _Float16* __restrict__ zh) {
  int i = blockIdx.x * blockDim.x + threadIdx.x;  // over groups of 4 floats
  float4 f = ((const float4*)z)[i];
  v4h h;
  h[0] = (_Float16)f.x; h[1] = (_Float16)f.y;
  h[2] = (_Float16)f.z; h[3] = (_Float16)f.w;
  ((v4h*)zh)[i] = h;
}

__global__ void cvt_w_kernel(const float* __restrict__ Wq, const float* __restrict__ Wk,
                             const float* __restrict__ Wv,
                             _Float16* __restrict__ qT, _Float16* __restrict__ kT,
                             _Float16* __restrict__ vT) {
  int e = blockIdx.x * blockDim.x + threadIdx.x;  // 0 .. 3*1024*128-1
  int w   = e >> 17;       // which matrix
  int rem = e & 131071;
  int kk  = rem >> 7;
  int n   = rem & 127;
  const float* src = (w == 0) ? Wq : (w == 1) ? Wk : Wv;
  _Float16*    dst = (w == 0) ? qT : (w == 1) ? kT : vT;
  dst[n * IN_D + kk] = (_Float16)src[kk * OUT_D + n];
}

// ------------------------------------------------------- fused QKV projection
// One wave per 16x16 output tile (same tile of q, k, v): reuse z fragment.
__global__ void qkv_kernel(const _Float16* __restrict__ zh,
                           const _Float16* __restrict__ wqT,
                           const _Float16* __restrict__ wkT,
                           const _Float16* __restrict__ wvT,
                           const float* __restrict__ bq, const float* __restrict__ bk,
                           const float* __restrict__ bv,
                           _Float16* __restrict__ qh, _Float16* __restrict__ kh,
                           _Float16* __restrict__ vT) {
  __shared__ __align__(16) _Float16 sm[4][16][16];
  int lane = threadIdx.x & 31;
  int wid  = threadIdx.x >> 5;
  int tile = blockIdx.x * 4 + wid;          // 4096 tiles total
  int mt = tile >> 3, nt = tile & 7;
  int m0 = mt * 16, n0 = nt * 16;
  int n = lane & 15, g = lane >> 4;

  v8f cq = zero8(), ck = zero8(), cv = zero8();
  const _Float16* za  = zh  + m0 * IN_D;
  const _Float16* bqp = wqT + n0 * IN_D;
  const _Float16* bkp = wkT + n0 * IN_D;
  const _Float16* bvp = wvT + n0 * IN_D;
  for (int kb = 0; kb < IN_D; kb += 32) {
    v16h a  = load_frag_rm(za,  IN_D, kb, lane);
    v16h b0 = load_frag_rm(bqp, IN_D, kb, lane);
    v16h b1 = load_frag_rm(bkp, IN_D, kb, lane);
    v16h b2 = load_frag_rm(bvp, IN_D, kb, lane);
    cq = __builtin_amdgcn_wmma_f32_16x16x32_f16(false, a, false, b0, (short)0, cq, false, false);
    ck = __builtin_amdgcn_wmma_f32_16x16x32_f16(false, a, false, b1, (short)0, ck, false, false);
    cv = __builtin_amdgcn_wmma_f32_16x16x32_f16(false, a, false, b2, (short)0, cv, false, false);
  }
  float bqv = bq[n0 + n], bkv = bk[n0 + n], bvv = bv[n0 + n];

  // q: bias + fold 1/sqrt(dh); stage C-layout -> row-major through LDS
#pragma unroll
  for (int r = 0; r < 8; ++r)
    sm[wid][r + 8 * g][n] = (_Float16)((cq[r] + bqv) * QSCALE);
  __syncthreads();
  {
    int rr = lane >> 1, seg = lane & 1;
    v8h row = *(const v8h*)&sm[wid][rr][seg * 8];
    *(v8h*)&qh[(m0 + rr) * OUT_D + n0 + seg * 8] = row;
  }
  __syncthreads();
  // k: same path
#pragma unroll
  for (int r = 0; r < 8; ++r)
    sm[wid][r + 8 * g][n] = (_Float16)(ck[r] + bkv);
  __syncthreads();
  {
    int rr = lane >> 1, seg = lane & 1;
    v8h row = *(const v8h*)&sm[wid][rr][seg * 8];
    *(v8h*)&kh[(m0 + rr) * OUT_D + n0 + seg * 8] = row;
  }
  // v: transposed store — C layout packs 8 consecutive M per lane -> one b128
  v8h vv;
#pragma unroll
  for (int r = 0; r < 8; ++r) vv[r] = (_Float16)(cv[r] + bvv);
  *(v8h*)&vT[(n0 + n) * NQ + m0 + 8 * g] = vv;
}

// -------------------------------------------------------- flash attention
// 4 waves/block, each wave owns 16 query rows. K/V tiles for the current 32
// keys are staged block-wide into LDS with async-to-LDS copies (double
// buffered), so all 4 waves share one fetch and global latency overlaps math.
__global__ void attn_kernel(const _Float16* __restrict__ qh,
                            const _Float16* __restrict__ kh,
                            const _Float16* __restrict__ vT,
                            float* __restrict__ out) {
  __shared__ __align__(16) _Float16 sk[2][32][OUT_D];  // K rows j0..j0+31      (2x8KB)
  __shared__ __align__(16) _Float16 sv[2][OUT_D][32];  // V^T cols, 32 keys     (2x8KB)
  __shared__ __align__(16) _Float16 sp[4][16][32];     // per-wave P staging    (4KB)

  int tid  = threadIdx.x;
  int lane = tid & 31;
  int wid  = tid >> 5;
  int mt = blockIdx.x * 4 + wid;            // 512 tiles
  int m0 = mt * 16;
  int n = lane & 15, g = lane >> 4;
  int m = n;                                 // A-fragment row index for this lane

  v16h aq[4];
#pragma unroll
  for (int t = 0; t < 4; ++t) aq[t] = load_frag_rm(qh + m0 * OUT_D, OUT_D, t * 32, lane);

  v8f o[8];
#pragma unroll
  for (int f = 0; f < 8; ++f) o[f] = zero8();
  v8f mrow, lrow;
#pragma unroll
  for (int r = 0; r < 8; ++r) { mrow[r] = -1e30f; lrow[r] = 0.0f; }

  // cooperative async stage of one 32-key K tile + V tile (8 b128 per thread)
  auto issue_tile = [&](int buf, int j) {
    unsigned kbase = (unsigned)(unsigned long long)&sk[buf][0][0];
    unsigned vbase = (unsigned)(unsigned long long)&sv[buf][0][0];
#pragma unroll
    for (int i = 0; i < 4; ++i) {           // K: 512 x 16B chunks
      int c = tid + 128 * i;
      int row = c >> 4, seg = c & 15;
      async_load_b128(kbase + c * 16, kh + (j + row) * OUT_D + seg * 8);
    }
#pragma unroll
    for (int i = 0; i < 4; ++i) {           // V^T: 512 x 16B chunks
      int c = tid + 128 * i;
      int vn = c >> 2, jseg = c & 3;
      async_load_b128(vbase + c * 16, vT + (size_t)vn * NQ + j + jseg * 8);
    }
  };

  issue_tile(0, 0);                          // prologue: buffer 0

  for (int it = 0; it < NQ / 32; ++it) {
    int j0  = it * 32;
    int cur = it & 1, nxt = cur ^ 1;
    int jn  = (j0 + 32) & (NQ - 1);          // wraps on last iter (harmless)
    issue_tile(nxt, jn);                     // overlap next fetch with compute
    wait_asynccnt_le8();                     // oldest 8 (current buffer) done
    __syncthreads();                         // visible block-wide

    // S = q @ k^T for 32 key columns (two 16-wide C tiles), K from LDS
    v8f slo = zero8(), shi = zero8();
#pragma unroll
    for (int t = 0; t < 4; ++t) {
      v16h blo = load_frag_rm(&sk[cur][0][0],  OUT_D, t * 32, lane);
      slo = __builtin_amdgcn_wmma_f32_16x16x32_f16(false, aq[t], false, blo, (short)0, slo, false, false);
      v16h bhi = load_frag_rm(&sk[cur][16][0], OUT_D, t * 32, lane);
      shi = __builtin_amdgcn_wmma_f32_16x16x32_f16(false, aq[t], false, bhi, (short)0, shi, false, false);
    }

    // online softmax: rows live across 16 lanes of each half-wave
    v8f alpha;
#pragma unroll
    for (int r = 0; r < 8; ++r) {
      float tmax = fmaxf(slo[r], shi[r]);
#pragma unroll
      for (int d = 1; d < 16; d <<= 1) tmax = fmaxf(tmax, __shfl_xor(tmax, d, 16));
      float mnew = fmaxf(mrow[r], tmax);
      float al = __expf(mrow[r] - mnew);
      float pl = __expf(slo[r] - mnew);
      float ph = __expf(shi[r] - mnew);
      float rs = pl + ph;
#pragma unroll
      for (int d = 1; d < 16; d <<= 1) rs += __shfl_xor(rs, d, 16);
      lrow[r] = lrow[r] * al + rs;
      mrow[r] = mnew;
      alpha[r] = al;
      // stage P (C layout) into LDS for A-layout reload
      sp[wid][r + 8 * g][n]      = (_Float16)pl;
      sp[wid][r + 8 * g][n + 16] = (_Float16)ph;
    }
    __syncthreads();
#pragma unroll
    for (int f = 0; f < 8; ++f)
#pragma unroll
      for (int r = 0; r < 8; ++r) o[f][r] *= alpha[r];
    FragA pa;
    const _Float16* lp = &sp[wid][0][0];
    pa.h[0] = *(const v8h*)(lp + m * 32 + 8 * g);
    pa.h[1] = *(const v8h*)(lp + m * 32 + 16 + 8 * g);

    // O += P @ V, V^T fragments from LDS
#pragma unroll
    for (int f = 0; f < 8; ++f) {
      v16h bv = load_frag_rm(&sv[cur][f * 16][0], 32, 0, lane);
      o[f] = __builtin_amdgcn_wmma_f32_16x16x32_f16(false, pa.v, false, bv, (short)0, o[f], false, false);
    }
    __syncthreads();                         // all reads of buffer `cur` + P done
  }

  // normalize and write fp32 output
#pragma unroll
  for (int r = 0; r < 8; ++r) {
    float inv = 1.0f / lrow[r];
#pragma unroll
    for (int f = 0; f < 8; ++f)
      out[(m0 + r + 8 * g) * OUT_D + f * 16 + n] = o[f][r] * inv;
  }
}

extern "C" void kernel_launch(void* const* d_in, const int* in_sizes, int n_in,
                              void* d_out, int out_size, void* d_ws, size_t ws_size,
                              hipStream_t stream) {
  const float* z  = (const float*)d_in[0];
  const float* Wq = (const float*)d_in[1];
  const float* bq = (const float*)d_in[2];
  const float* Wk = (const float*)d_in[3];
  const float* bk = (const float*)d_in[4];
  const float* Wv = (const float*)d_in[5];
  const float* bv = (const float*)d_in[6];
  float* out = (float*)d_out;

  char* ws = (char*)d_ws;
  _Float16* zh  = (_Float16*)(ws);                                   // 16 MB
  _Float16* wqT = (_Float16*)(ws + (size_t)16 * 1024 * 1024);        // 256 KB each
  _Float16* wkT = wqT + 128 * 1024;
  _Float16* wvT = wkT + 128 * 1024;
  _Float16* qh  = wvT + 128 * 1024;                                  // 2 MB each
  _Float16* kh  = qh + (size_t)NQ * OUT_D;
  _Float16* vT  = kh + (size_t)NQ * OUT_D;

  cvt_z_kernel<<<(NQ * IN_D / 4) / 256, 256, 0, stream>>>(z, zh);
  cvt_w_kernel<<<(3 * IN_D * OUT_D) / 256, 256, 0, stream>>>(Wq, Wk, Wv, wqT, wkT, wvT);
  qkv_kernel<<<1024, 128, 0, stream>>>(zh, wqT, wkT, wvT, bq, bk, bv, qh, kh, vT);
  attn_kernel<<<128, 128, 0, stream>>>(qh, kh, vT, out);
}
